// MultiChannelGNNBlock_27084063768606
// MI455X (gfx1250) — matrix operand order repaired and checked by reading, not compiled, not run on hardware
//
#include <hip/hip_runtime.h>
#include <hip/hip_bf16.h>
#include <math.h>

#define D   128
#define NN  40000
#define BG  8
#define SG  5000
#define KNN 6
#define EE  640000
#define FF  256

typedef __attribute__((ext_vector_type(16))) __bf16 bf16x16;
typedef __attribute__((ext_vector_type(8)))  __bf16 bf16x8;
typedef __attribute__((ext_vector_type(8)))  float  f32x8;

// ---------------------------------------------------------------------------
// WMMA helpers (CDNA5 wave32, V_WMMA_F32_16X16X32_BF16)
// ---------------------------------------------------------------------------
__device__ __forceinline__ f32x8 wmma_bf16(bf16x16 a, bf16x16 b, f32x8 c) {
  // (neg_a, A, neg_b, B, c_mod, C, reuse_a, reuse_b)
  return __builtin_amdgcn_wmma_f32_16x16x32_bf16(false, a, false, b, (short)0, c,
                                                 false, false);
}

__device__ __forceinline__ bf16x16 cat8(bf16x8 lo, bf16x8 hi) {
  return __builtin_shufflevector(lo, hi, 0,1,2,3,4,5,6,7,8,9,10,11,12,13,14,15);
}

// A fragment 16x32 (MxK) from row-major bf16 (leading dim ld).
// Lane l holds row M=l&15, K in {16h..16h+15}, h=l>>4 -> two 16B loads.
__device__ __forceinline__ bf16x16 load_a_frag(const __bf16* A, int ld, int mrow,
                                               int k0, int lane) {
  int h = lane >> 4;
  const __bf16* p = A + (size_t)mrow * ld + k0 + 16 * h;
  bf16x8 lo = *(const bf16x8*)p;
  bf16x8 hi = *(const bf16x8*)(p + 8);
  return cat8(lo, hi);
}

// B fragment 32x16 (KxN) where B = X^T, X row-major: column n of B is row n of X.
// Lane l holds N=l&15, K = k0 + 16h + e -> one contiguous 32B load.
__device__ __forceinline__ bf16x16 load_bT_frag(const __bf16* X, int ld, int nrow,
                                                int k0, int lane) {
  int h = lane >> 4;
  return *(const bf16x16*)(X + (size_t)nrow * ld + k0 + 16 * h);
}

// B fragment from pre-packed weights: contiguous 32B per (ktile, ntile, lane).
__device__ __forceinline__ bf16x16 load_bp_frag(const __bf16* Bp, int ntiles,
                                                int kt, int nt, int lane) {
  return *(const bf16x16*)(Bp + ((size_t)(kt * ntiles + nt) * 32 + lane) * 16);
}

// ---------------------------------------------------------------------------
// Pack a row-major f32 weight [K x Nn] into WMMA B-fragment order (bf16).
// ---------------------------------------------------------------------------
__global__ __launch_bounds__(256)
void pack_b_kernel(const float* __restrict__ W, __bf16* __restrict__ out,
                   int K, int Nn) {
  int t = blockIdx.x * 256 + threadIdx.x;
  if (t >= K * Nn) return;
  int e    = t & 15;
  int lane = (t >> 4) & 31;
  int ntiles = Nn >> 4;
  int nt   = (t >> 9) % ntiles;
  int kt   = t / (512 * ntiles);
  int k = kt * 32 + 16 * (lane >> 4) + e;
  int n = nt * 16 + (lane & 15);
  out[t] = (__bf16)W[(size_t)k * Nn + n];
}

// ---------------------------------------------------------------------------
// Fused WMMA GEMM: C = act(A(bf16)[MxK] @ Bp + bias + resid)
// One wave computes a 16x64 strip: 4 N-tiles reuse one A fragment per k-step,
// 4 independent accumulators break the WMMA->WMMA RAW chain.
// grid(M/16, Nn/64), block 32.
// ---------------------------------------------------------------------------
__global__ __launch_bounds__(32)
void gemm_bf16_kernel(const __bf16* __restrict__ A, const __bf16* __restrict__ Bp,
                      const float* __restrict__ bias, const float* __restrict__ resid,
                      float* __restrict__ Cf, __bf16* __restrict__ Cbf,
                      int K, int Nn, int act) {
  int lane = threadIdx.x;
  int mt  = blockIdx.x;
  int nt4 = blockIdx.y;
  int ntiles = Nn >> 4;
  int h = lane >> 4;
  int mrow = mt * 16 + (lane & 15);
  f32x8 c[4];
#pragma unroll
  for (int t = 0; t < 4; ++t) {
    float bv = bias ? bias[(nt4 * 4 + t) * 16 + (lane & 15)] : 0.0f;
    c[t] = (f32x8){bv, bv, bv, bv, bv, bv, bv, bv};
  }
  for (int kt = 0; kt < (K >> 5); ++kt) {
    bf16x16 a = load_a_frag(A, K, mrow, kt * 32, lane);
#pragma unroll
    for (int t = 0; t < 4; ++t) {
      bf16x16 b = load_bp_frag(Bp, ntiles, kt, nt4 * 4 + t, lane);
      c[t] = wmma_bf16(a, b, c[t]);
    }
  }
#pragma unroll
  for (int t = 0; t < 4; ++t) {
    int n = (nt4 * 4 + t) * 16 + (lane & 15);
#pragma unroll
    for (int r = 0; r < 8; ++r) {
      int m = mt * 16 + 8 * h + r;
      float v = c[t][r];
      if (resid) v += resid[(size_t)m * Nn + n];
      if (act)   v = fminf(fmaxf(v, 0.0f), 6.0f);
      if (Cf)    Cf[(size_t)m * Nn + n] = v;
      if (Cbf)   Cbf[(size_t)m * Nn + n] = (__bf16)v;
    }
  }
}

// ---------------------------------------------------------------------------
// Casts / norms
// ---------------------------------------------------------------------------
__global__ __launch_bounds__(256)
void cast_bf16_kernel(const float* __restrict__ x, __bf16* __restrict__ xb, int n) {
  int t = blockIdx.x * 256 + threadIdx.x;
  if (t < n) xb[t] = (__bf16)x[t];
}

__global__ __launch_bounds__(256)
void sqnorm_kernel(const float* __restrict__ x, float* __restrict__ sq) {
  int i = blockIdx.x * 256 + threadIdx.x;
  if (i >= NN) return;
  float s = 0.0f;
  for (int d = 0; d < D; ++d) { float v = x[(size_t)i * D + d]; s += v * v; }
  sq[i] = s;
}

// ---------------------------------------------------------------------------
// kNN: per graph, Gram matrix via WMMA, per-row running top-6 of
// (sq[j] - 2*G[i][j]).  Block = 4 waves = 128 query rows.  Each wave keeps its
// 8 A-fragments (32 rows x K=128) resident in VGPRs for the whole scan; the
// shared 16-row column tile is double-buffered in LDS via CDNA5 async-to-LDS
// DMA (global_load_async_to_lds_b128 + s_wait_asynccnt).
// grid(ceil(SG/128)=40, BG), block 128.
// ---------------------------------------------------------------------------
#define NT_TILES 313  // ceil(5000/16)
__global__ __launch_bounds__(128)
void knn_kernel(const __bf16* __restrict__ xbf, const float* __restrict__ sqn,
                int* __restrict__ knn) {
  __shared__ __attribute__((aligned(64))) __bf16 bstage[2][16 * D]; // 2 x 4KB
  __shared__ float sc[4][32][17];

  int g = blockIdx.y;
  int tid = threadIdx.x;
  int lane = tid & 31, wv = tid >> 5;
  int gb = g * SG;
  const __bf16* Xg = xbf + (size_t)gb * D;
  int m0 = blockIdx.x * 128 + wv * 32;

  // Loop-invariant A fragments: this wave's 32 query rows, all 4 k-steps.
  int ar0 = m0 + (lane & 15);      if (ar0 >= SG) ar0 = 0;
  int ar1 = m0 + 16 + (lane & 15); if (ar1 >= SG) ar1 = 0;
  bf16x16 a0[4], a1[4];
#pragma unroll
  for (int ks = 0; ks < 4; ++ks) {
    a0[ks] = load_a_frag(Xg, D, ar0, ks * 32, lane);
    a1[ks] = load_a_frag(Xg, D, ar1, ks * 32, lane);
  }

  float bd[6]; int bi[6];
#pragma unroll
  for (int q = 0; q < 6; ++q) { bd[q] = 1e30f; bi[q] = 0; }

  int h = lane >> 4;
  int myrow = m0 + lane;

  // Stage column tile nt (16 rows x 128 cols bf16 = 4KB) into LDS buffer p.
  // 256 x 16B chunks; each of the 128 threads issues 2 async B128 copies.
  auto stage = [&](int nt, int p) {
    unsigned base = (unsigned)(uintptr_t)(&bstage[p][0]);
#pragma unroll
    for (int cch = 0; cch < 2; ++cch) {
      int ch = tid + cch * 128;            // 0..255
      int r  = ch >> 4;                    // row within tile
      int jn = nt * 16 + r;
      int jr = (jn < SG) ? jn : 0;         // clamp; masked later via jv
      unsigned lds_off = base + (unsigned)ch * 16u;
      unsigned long long gsrc =
          (unsigned long long)(uintptr_t)(Xg + (size_t)jr * D + (ch & 15) * 8);
      asm volatile("global_load_async_to_lds_b128 %0, %1, off"
                   :: "v"(lds_off), "v"(gsrc) : "memory");
    }
  };

  stage(0, 0);
  for (int nt = 0; nt < NT_TILES; ++nt) {
    int p = nt & 1;
    int nxt = (nt + 1 < NT_TILES) ? nt + 1 : nt;  // redundant last restage, drained below
    stage(nxt, p ^ 1);
    asm volatile("s_wait_asynccnt 2" ::: "memory");  // current tile's copies done
    __syncthreads();                                  // visible to all waves

    f32x8 c0 = {0,0,0,0,0,0,0,0};
    f32x8 c1 = {0,0,0,0,0,0,0,0};
#pragma unroll
    for (int ks = 0; ks < 4; ++ks) {
      bf16x16 b = *(const bf16x16*)&bstage[p][(lane & 15) * D + ks * 32 + 16 * h];
      c0 = wmma_bf16(a0[ks], b, c0);
      c1 = wmma_bf16(a1[ks], b, c1);
    }

    int jn = nt * 16 + (lane & 15);
    int jr = (jn < SG) ? jn : 0;
    float sqj = sqn[gb + jr];
    bool jv = (jn < SG);
#pragma unroll
    for (int r = 0; r < 8; ++r) {
      sc[wv][8 * h + r][lane & 15]      = jv ? (sqj - 2.0f * c0[r]) : 1e30f;
      sc[wv][16 + 8 * h + r][lane & 15] = jv ? (sqj - 2.0f * c1[r]) : 1e30f;
    }
    __syncthreads();

#pragma unroll
    for (int q2 = 0; q2 < 16; ++q2) {
      int j = nt * 16 + q2;
      float v = sc[wv][lane][q2];
      if (j == myrow) v = 1e30f;           // exclude self
      if (v < bd[5]) {                     // cheap reject; rare insert
        float nv = v; int nj = j;
#pragma unroll
        for (int q = 0; q < 6; ++q) {
          bool lt = nv < bd[q];
          float tv = lt ? bd[q] : nv; int tj = lt ? bi[q] : nj;
          bd[q] = lt ? nv : bd[q];    bi[q] = lt ? nj : bi[q];
          nv = tv; nj = tj;
        }
      }
    }
  }
  asm volatile("s_wait_asynccnt 0" ::: "memory");     // drain redundant restage

  if (myrow < SG) {
#pragma unroll
    for (int q = 0; q < 6; ++q)
      knn[(size_t)(gb + myrow) * KNN + q] = gb + bi[q];
  }
}

// ---------------------------------------------------------------------------
// Latent GATv2: regular degree (6 knn + self loop) -> one wave per node.
// Writes ch_lat (bf16) into cat[:, 128:256].
// ---------------------------------------------------------------------------
__global__ __launch_bounds__(128)
void latent_gat_kernel(const float* __restrict__ xl, const float* __restrict__ xr,
                       const float* __restrict__ att, const float* __restrict__ bias,
                       const int* __restrict__ knn, __bf16* __restrict__ catb) {
  int node = blockIdx.x * 4 + threadIdx.y;
  if (node >= NN) return;
  int lane = threadIdx.x;
  int srcs[7];
#pragma unroll
  for (int q = 0; q < 6; ++q) srcs[q] = knn[(size_t)node * KNN + q];
  srcs[6] = node;
  float xrv[4], attv[4], xlv[7][4], logit[7];
#pragma unroll
  for (int d4 = 0; d4 < 4; ++d4) {
    int d = lane + 32 * d4;
    xrv[d4]  = xr[(size_t)node * D + d];
    attv[d4] = att[d];
  }
#pragma unroll
  for (int j = 0; j < 7; ++j) {
    float acc = 0.0f;
#pragma unroll
    for (int d4 = 0; d4 < 4; ++d4) {
      int d = lane + 32 * d4;
      float xv = xl[(size_t)srcs[j] * D + d];
      xlv[j][d4] = xv;
      float hh = xv + xrv[d4];
      hh = (hh > 0.0f) ? hh : 0.2f * hh;
      acc += hh * attv[d4];
    }
#pragma unroll
    for (int off = 16; off > 0; off >>= 1) acc += __shfl_xor(acc, off, 32);
    logit[j] = acc;
  }
  float m = logit[0];
#pragma unroll
  for (int j = 1; j < 7; ++j) m = fmaxf(m, logit[j]);
  float s = 0.0f, al[7];
#pragma unroll
  for (int j = 0; j < 7; ++j) { al[j] = __expf(logit[j] - m); s += al[j]; }
  float inv = 1.0f / s;
#pragma unroll
  for (int d4 = 0; d4 < 4; ++d4) {
    int d = lane + 32 * d4;
    float o = bias[d];
#pragma unroll
    for (int j = 0; j < 7; ++j) o += al[j] * inv * xlv[j][d4];
    catb[(size_t)node * (2 * D) + D + d] = (__bf16)o;
  }
}

// ---------------------------------------------------------------------------
// Spatial GATv2 (irregular edges): 3-pass segment softmax with atomics.
// ---------------------------------------------------------------------------
__device__ __forceinline__ unsigned fkey(float f) {
  unsigned u = __float_as_uint(f);
  return (u & 0x80000000u) ? ~u : (u | 0x80000000u);
}
__device__ __forceinline__ float fdecode(unsigned k) {
  unsigned u = (k & 0x80000000u) ? (k & 0x7FFFFFFFu) : ~k;
  return __uint_as_float(u);
}

__global__ __launch_bounds__(128)
void sp_logit_kernel(const long long* __restrict__ ei, const float* __restrict__ ea,
                     const float* __restrict__ xl, const float* __restrict__ xr,
                     const float* __restrict__ att, const float* __restrict__ wse,
                     float* __restrict__ wbuf, unsigned* __restrict__ mkey) {
  int e = blockIdx.x * 4 + threadIdx.y;
  if (e >= EE) return;
  int lane = threadIdx.x;
  int src = (int)ei[e], dst = (int)ei[EE + e];
  float a = ea[e];
  float acc = 0.0f;
#pragma unroll
  for (int d4 = 0; d4 < 4; ++d4) {
    int d = lane + 32 * d4;
    float hh = xl[(size_t)src * D + d] + xr[(size_t)dst * D + d] + a * wse[d];
    hh = (hh > 0.0f) ? hh : 0.2f * hh;
    acc += hh * att[d];
  }
#pragma unroll
  for (int off = 16; off > 0; off >>= 1) acc += __shfl_xor(acc, off, 32);
  if (lane == 0) {
    wbuf[e] = acc;
    atomicMax(&mkey[dst], fkey(acc));
  }
}

__global__ __launch_bounds__(256)
void sp_expsum_kernel(const long long* __restrict__ ei, float* __restrict__ wbuf,
                      const unsigned* __restrict__ mkey, float* __restrict__ sb) {
  int e = blockIdx.x * 256 + threadIdx.x;
  if (e >= EE) return;
  int dst = (int)ei[EE + e];
  float ex = __expf(wbuf[e] - fdecode(mkey[dst]));
  wbuf[e] = ex;
  atomicAdd(&sb[dst], ex);
}

__global__ __launch_bounds__(128)
void sp_scatter_kernel(const long long* __restrict__ ei, const float* __restrict__ wbuf,
                       const float* __restrict__ sb, const float* __restrict__ xl,
                       float* __restrict__ accb) {
  int e = blockIdx.x * 4 + threadIdx.y;
  if (e >= EE) return;
  int lane = threadIdx.x;
  int src = (int)ei[e], dst = (int)ei[EE + e];
  float alpha = wbuf[e] / sb[dst];
#pragma unroll
  for (int d4 = 0; d4 < 4; ++d4) {
    int d = lane + 32 * d4;
    atomicAdd(&accb[(size_t)dst * D + d], alpha * xl[(size_t)src * D + d]);
  }
}

__global__ __launch_bounds__(256)
void sp_finalize_kernel(const float* __restrict__ accb, const float* __restrict__ bias,
                        __bf16* __restrict__ catb) {
  int t = blockIdx.x * 256 + threadIdx.x;
  if (t >= NN * D) return;
  int i = t >> 7, d = t & 127;
  catb[(size_t)i * (2 * D) + d] = (__bf16)(accb[t] + bias[d]);
}

// ---------------------------------------------------------------------------
// Host launcher
// ---------------------------------------------------------------------------
extern "C" void kernel_launch(void* const* d_in, const int* in_sizes, int n_in,
                              void* d_out, int out_size, void* d_ws, size_t ws_size,
                              hipStream_t stream) {
  const float*     x      = (const float*)d_in[0];
  const long long* ei     = (const long long*)d_in[1];
  const float*     ea     = (const float*)d_in[2];
  const float*     W_sl   = (const float*)d_in[4];
  const float*     b_sl   = (const float*)d_in[5];
  const float*     W_sr   = (const float*)d_in[6];
  const float*     b_sr   = (const float*)d_in[7];
  const float*     att_s  = (const float*)d_in[8];
  const float*     W_se   = (const float*)d_in[9];
  const float*     bias_s = (const float*)d_in[10];
  const float*     W_ll   = (const float*)d_in[11];
  const float*     b_ll   = (const float*)d_in[12];
  const float*     W_lr   = (const float*)d_in[13];
  const float*     b_lr   = (const float*)d_in[14];
  const float*     att_l  = (const float*)d_in[15];
  const float*     bias_l = (const float*)d_in[16];
  const float*     W_c    = (const float*)d_in[17];
  const float*     b_c    = (const float*)d_in[18];
  const float*     W_f1   = (const float*)d_in[19];
  const float*     b_f1   = (const float*)d_in[20];
  const float*     W_f2   = (const float*)d_in[21];
  const float*     b_f2   = (const float*)d_in[22];
  float* out = (float*)d_out;

  char* w = (char*)d_ws;
  auto alloc = [&](size_t bytes) -> char* {
    char* p = w;
    w += (bytes + 255) & ~(size_t)255;
    return p;
  };

  // atomics region (contiguous -> single memset)
  unsigned* mkey = (unsigned*)alloc((size_t)NN * 4);
  float*    sb   = (float*)   alloc((size_t)NN * 4);
  float*    accb = (float*)   alloc((size_t)NN * D * 4);
  size_t zero_bytes = (size_t)NN * 4 + (size_t)NN * 4 + (size_t)NN * D * 4;

  __bf16* xbf  = (__bf16*)alloc((size_t)NN * D * 2);
  float*  sqn  = (float*) alloc((size_t)NN * 4);
  __bf16* P_sl = (__bf16*)alloc((size_t)D * D * 2);
  __bf16* P_sr = (__bf16*)alloc((size_t)D * D * 2);
  __bf16* P_ll = (__bf16*)alloc((size_t)D * D * 2);
  __bf16* P_lr = (__bf16*)alloc((size_t)D * D * 2);
  __bf16* P_c  = (__bf16*)alloc((size_t)2 * D * D * 2);
  __bf16* P_f1 = (__bf16*)alloc((size_t)D * FF * 2);
  __bf16* P_f2 = (__bf16*)alloc((size_t)FF * D * 2);
  float*  xls  = (float*) alloc((size_t)NN * D * 4);
  float*  xrs  = (float*) alloc((size_t)NN * D * 4);
  float*  xll  = (float*) alloc((size_t)NN * D * 4);
  float*  xlr  = (float*) alloc((size_t)NN * D * 4);
  int*    knnb = (int*)   alloc((size_t)NN * KNN * 4);
  float*  wbuf = (float*) alloc((size_t)EE * 4);
  __bf16* catb = (__bf16*)alloc((size_t)NN * 2 * D * 2);
  float*  fusedf = (float*) alloc((size_t)NN * D * 4);
  __bf16* fusedb = (__bf16*)alloc((size_t)NN * D * 2);
  __bf16* hbf    = (__bf16*)alloc((size_t)NN * FF * 2);

  hipMemsetAsync(mkey, 0, zero_bytes, stream);

  // casts + norms
  cast_bf16_kernel<<<(NN * D + 255) / 256, 256, 0, stream>>>(x, xbf, NN * D);
  sqnorm_kernel<<<(NN + 255) / 256, 256, 0, stream>>>(x, sqn);

  // weight packing (tiny)
  pack_b_kernel<<<(D * D + 255) / 256, 256, 0, stream>>>(W_sl, P_sl, D, D);
  pack_b_kernel<<<(D * D + 255) / 256, 256, 0, stream>>>(W_sr, P_sr, D, D);
  pack_b_kernel<<<(D * D + 255) / 256, 256, 0, stream>>>(W_ll, P_ll, D, D);
  pack_b_kernel<<<(D * D + 255) / 256, 256, 0, stream>>>(W_lr, P_lr, D, D);
  pack_b_kernel<<<(2 * D * D + 255) / 256, 256, 0, stream>>>(W_c, P_c, 2 * D, D);
  pack_b_kernel<<<(D * FF + 255) / 256, 256, 0, stream>>>(W_f1, P_f1, D, FF);
  pack_b_kernel<<<(FF * D + 255) / 256, 256, 0, stream>>>(W_f2, P_f2, FF, D);

  // node transforms (WMMA GEMM): xl/xr for both channels
  dim3 gA(NN / 16, D / 64);
  gemm_bf16_kernel<<<gA, 32, 0, stream>>>(xbf, P_sl, b_sl, nullptr, xls, nullptr, D, D, 0);
  gemm_bf16_kernel<<<gA, 32, 0, stream>>>(xbf, P_sr, b_sr, nullptr, xrs, nullptr, D, D, 0);
  gemm_bf16_kernel<<<gA, 32, 0, stream>>>(xbf, P_ll, b_ll, nullptr, xll, nullptr, D, D, 0);
  gemm_bf16_kernel<<<gA, 32, 0, stream>>>(xbf, P_lr, b_lr, nullptr, xlr, nullptr, D, D, 0);

  // dynamic kNN graph (WMMA Gram + async-LDS double buffering + top-6)
  knn_kernel<<<dim3((SG + 127) / 128, BG), 128, 0, stream>>>(xbf, sqn, knnb);

  // latent channel -> cat[:, 128:256]
  latent_gat_kernel<<<NN / 4, dim3(32, 4), 0, stream>>>(xll, xlr, att_l, bias_l,
                                                        knnb, catb);

  // spatial channel -> cat[:, 0:128]
  sp_logit_kernel<<<EE / 4, dim3(32, 4), 0, stream>>>(ei, ea, xls, xrs, att_s,
                                                      W_se, wbuf, mkey);
  sp_expsum_kernel<<<(EE + 255) / 256, 256, 0, stream>>>(ei, wbuf, mkey, sb);
  sp_scatter_kernel<<<EE / 4, dim3(32, 4), 0, stream>>>(ei, wbuf, sb, xls, accb);
  sp_finalize_kernel<<<(NN * D + 255) / 256, 256, 0, stream>>>(accb, bias_s, catb);

  // fusion: fused = cat @ W_c + b_c + x
  gemm_bf16_kernel<<<dim3(NN / 16, D / 64), 32, 0, stream>>>(catb, P_c, b_c, x,
                                                             fusedf, fusedb, 2 * D, D, 0);
  // FFN1: h = relu6(fused @ W_f1 + b_f1)
  gemm_bf16_kernel<<<dim3(NN / 16, FF / 64), 32, 0, stream>>>(fusedb, P_f1, b_f1,
                                                              nullptr, nullptr, hbf,
                                                              D, FF, 1);
  // FFN2: out = relu6(fused + h @ W_f2 + b_f2)
  gemm_bf16_kernel<<<dim3(NN / 16, D / 64), 32, 0, stream>>>(hbf, P_f2, b_f2,
                                                             fusedf, out, nullptr,
                                                             FF, D, 1);
}
